// CONPool_62680752717909
// MI455X (gfx1250) — compile-verified, matrix operand order checked by reading, not run on hardware
//
#include <hip/hip_runtime.h>
#include <hip/hip_bf16.h>
#include <math.h>
#include <stdint.h>

typedef __attribute__((ext_vector_type(16))) _Float16 v16h;
typedef __attribute__((ext_vector_type(8)))  _Float16 h8;
typedef __attribute__((ext_vector_type(8)))  float    v8f;

#define NN 50000
#define EE 800000
#define DD 128
#define KK 40000
#define NB_N 196          /* ceil(N/256) */
#define NEG_SLOPE 0.2f

// ---------------------------------------------------------------------------
// small utility kernels
// ---------------------------------------------------------------------------
__global__ void fill_f32(float* p, float v, int n) {
  int i = blockIdx.x * blockDim.x + threadIdx.x;
  if (i < n) p[i] = v;
}
__global__ void fill_i32(int* p, int v, int n) {
  int i = blockIdx.x * blockDim.x + threadIdx.x;
  if (i < n) p[i] = v;
}
__global__ void fill_u32(unsigned* p, unsigned v, int n) {
  int i = blockIdx.x * blockDim.x + threadIdx.x;
  if (i < n) p[i] = v;
}

// transpose + fp32->f16 pack of [W_rel ; W_root] into B^T layout [o][k], k=0..255
__global__ void convw_k(const float* __restrict__ Wrel,
                        const float* __restrict__ Wroot,
                        _Float16* __restrict__ outT) {
  int idx = blockIdx.x * blockDim.x + threadIdx.x;   // 0..32767
  if (idx >= 128 * 256) return;
  int o = idx >> 8;
  int i = idx & 255;
  float v = (i < 128) ? Wrel[i * 128 + o] : Wroot[(i - 128) * 128 + o];
  outT[o * 256 + i] = (_Float16)v;
}

// ---------------------------------------------------------------------------
// edge scatter: agg[dst] += x[src]  (wave per edge, float4 per lane)
// ---------------------------------------------------------------------------
__global__ void scatter_rows_k(const float* __restrict__ xin,
                               const int* __restrict__ src,
                               const int* __restrict__ dst,
                               float* __restrict__ agg, int nE) {
  int gid = blockIdx.x * blockDim.x + threadIdx.x;
  int e = gid >> 5;
  int lane = gid & 31;
  if (e >= nE) return;
  int s = src[e], d = dst[e];
  float4 v = ((const float4*)(xin + (size_t)s * DD))[lane];
  float* o = agg + (size_t)d * DD + lane * 4;
  atomicAdd(o + 0, v.x);
  atomicAdd(o + 1, v.y);
  atomicAdd(o + 2, v.z);
  atomicAdd(o + 3, v.w);
}

// scalar edge scatter: sagg[dst] += y[src]
__global__ void scatter_scalar_k(const float* __restrict__ y,
                                 const int* __restrict__ src,
                                 const int* __restrict__ dst,
                                 float* __restrict__ sagg, int nE) {
  int e = blockIdx.x * blockDim.x + threadIdx.x;
  if (e >= nE) return;
  atomicAdd(&sagg[dst[e]], y[src[e]]);
}

// ---------------------------------------------------------------------------
// fused GraphConv GEMM:  C = act( [agg|x] (N x 256) @ B (256 x 128) + bias )
// WMMA f32_16x16x32_f16, B^T staged in LDS (64 KB) via CDNA5
// GLOBAL_LOAD_ASYNC_TO_LDS_B128 (ASYNCcnt-tracked, bypasses VGPRs),
// 8 waves/block, each wave computes a 16 x 128 tile (8 accumulators).
// ---------------------------------------------------------------------------
__global__ __launch_bounds__(256, 1)
void gemm_fused_k(const float* __restrict__ Aagg,
                  const float* __restrict__ Aroot,
                  const _Float16* __restrict__ wT,   // [128][256] f16
                  const float* __restrict__ bias,    // [128]
                  float* __restrict__ Cout,          // [N][128]
                  int n, int applyRelu) {
  __shared__ _Float16 ldsW[128 * 256];               // 64 KB
  int tid = threadIdx.x;

  // --- async stage of B^T into LDS: 4096 x 16B chunks, 16 per thread ------
  {
    unsigned ldsBase = (unsigned)(uintptr_t)&ldsW[0];      // flat->LDS: addr[31:0]
    const char* gsrc = (const char*)wT;
    #pragma unroll
    for (int j = tid; j < 4096; j += 256) {
      unsigned laddr = ldsBase + (unsigned)j * 16u;
      unsigned long long gaddr = (unsigned long long)(uintptr_t)(gsrc + (size_t)j * 16);
      asm volatile("global_load_async_to_lds_b128 %0, %1, off"
                   :: "v"(laddr), "v"(gaddr) : "memory");
    }
    asm volatile("s_wait_asynccnt 0x0" ::: "memory");
  }
  __syncthreads();

  int wave = tid >> 5;
  int lane = tid & 31;
  int rowBase = blockIdx.x * 128 + wave * 16;
  int r = rowBase + (lane & 15);                     // A-operand row for this lane
  int kbase = (lane >> 4) * 8;                       // 0 or 8 (WMMA A layout)

  if (r < n) {                                       // warm L2/WGP$ for A rows
    __builtin_prefetch(Aagg + (size_t)r * DD, 0, 0);
    __builtin_prefetch(Aroot + (size_t)r * DD, 0, 0);
  }

  v8f acc[8] = {};

  #pragma unroll
  for (int ks = 0; ks < 8; ++ks) {                   // K = 256 in steps of 32
    const float* Asrc = (ks < 4) ? Aagg : Aroot;
    int kcol = (ks & 3) * 32 + kbase;

    v16h a;
    if (r < n) {
      const float* rp = Asrc + (size_t)r * DD + kcol;
      float4 f0 = *(const float4*)(rp + 0);
      float4 f1 = *(const float4*)(rp + 4);
      float4 f2 = *(const float4*)(rp + 16);
      float4 f3 = *(const float4*)(rp + 20);
      a[0] = (_Float16)f0.x;  a[1] = (_Float16)f0.y;
      a[2] = (_Float16)f0.z;  a[3] = (_Float16)f0.w;
      a[4] = (_Float16)f1.x;  a[5] = (_Float16)f1.y;
      a[6] = (_Float16)f1.z;  a[7] = (_Float16)f1.w;
      a[8] = (_Float16)f2.x;  a[9] = (_Float16)f2.y;
      a[10] = (_Float16)f2.z; a[11] = (_Float16)f2.w;
      a[12] = (_Float16)f3.x; a[13] = (_Float16)f3.y;
      a[14] = (_Float16)f3.z; a[15] = (_Float16)f3.w;
    } else {
      #pragma unroll
      for (int i = 0; i < 16; ++i) a[i] = (_Float16)0.0f;
    }

    #pragma unroll
    for (int c = 0; c < 8; ++c) {                    // 8 column tiles of 16
      int o = c * 16 + (lane & 15);                  // B column = row of B^T
      const _Float16* bp = ldsW + o * 256 + ks * 32 + kbase;
      h8 lo = *(const h8*)bp;                        // 16B LDS load
      h8 hi = *(const h8*)(bp + 16);                 // 16B LDS load
      v16h b;
      #pragma unroll
      for (int i = 0; i < 8; ++i) { b[i] = lo[i]; b[8 + i] = hi[i]; }
      acc[c] = __builtin_amdgcn_wmma_f32_16x16x32_f16(
          /*neg_a=*/false, a, /*neg_b=*/false, b,
          /*c_mod=*/(short)0, acc[c], /*reuse_a=*/false, /*reuse_b=*/false);
    }
  }

  // epilogue: C/D layout — lane gives col (N=lane&15), VGPR v gives row M=v(+8)
  int ncol0 = lane & 15;
  int mAdd = (lane >> 4) * 8;
  #pragma unroll
  for (int c = 0; c < 8; ++c) {
    int col = c * 16 + ncol0;
    float bv = bias[col];
    #pragma unroll
    for (int v = 0; v < 8; ++v) {
      int g = rowBase + v + mAdd;
      if (g < n) {
        float val = acc[c][v] + bv;
        if (applyRelu) val = (val > 0.0f) ? val : NEG_SLOPE * val;
        Cout[(size_t)g * DD + col] = val;
      }
    }
  }
}

// ---------------------------------------------------------------------------
// dual dot-products: y1 = x.w1, y2 = x.w2 (wave per node)
// ---------------------------------------------------------------------------
__global__ void dots2_k(const float* __restrict__ xin,
                        const float* __restrict__ w1,
                        const float* __restrict__ w2,
                        float* __restrict__ y1, float* __restrict__ y2, int n) {
  int gid = blockIdx.x * blockDim.x + threadIdx.x;
  int node = gid >> 5;
  int lane = gid & 31;
  if (node >= n) return;
  float4 a  = ((const float4*)(xin + (size_t)node * DD))[lane];
  float4 b1 = ((const float4*)w1)[lane];
  float4 b2 = ((const float4*)w2)[lane];
  float p1 = a.x * b1.x + a.y * b1.y + a.z * b1.z + a.w * b1.w;
  float p2 = a.x * b2.x + a.y * b2.y + a.z * b2.z + a.w * b2.w;
  #pragma unroll
  for (int off = 16; off; off >>= 1) {
    p1 += __shfl_xor(p1, off, 32);
    p2 += __shfl_xor(p2, off, 32);
  }
  if (lane == 0) { y1[node] = p1; y2[node] = p2; }
}

__global__ void combine_s_k(const float* sagg, const float* bias1,
                            const float* yroot, float* sout, int n) {
  int i = blockIdx.x * blockDim.x + threadIdx.x;
  if (i < n) sout[i] = sagg[i] + bias1[0] + yroot[i];
}

__global__ void mean2_k(const float* a, const float* b, float* out, int n) {
  int i = blockIdx.x * blockDim.x + threadIdx.x;
  if (i < n) out[i] = 0.5f * (a[i] + b[i]);
}

// ---------------------------------------------------------------------------
// deterministic top-K select (radix-histogram threshold + ranked compaction)
// ---------------------------------------------------------------------------
__device__ __forceinline__ unsigned keybits(float f) {
  unsigned u = __float_as_uint(f);
  return (u & 0x80000000u) ? ~u : (u | 0x80000000u);   // descending f == descending key
}

__global__ void hist_k(const float* __restrict__ keys, int n, unsigned* hist) {
  int i = blockIdx.x * blockDim.x + threadIdx.x;
  if (i < n) atomicAdd(&hist[keybits(keys[i]) >> 20], 1u);
}

__global__ void thresh_k(const unsigned* hist, unsigned* tinfo, int Kw) {
  if (threadIdx.x | blockIdx.x) return;
  unsigned cum = 0; unsigned t = 0;
  for (int b = 4095; b >= 0; --b) {
    unsigned c = hist[b];
    if (cum + c >= (unsigned)Kw) { t = (unsigned)b; break; }
    cum += c;
  }
  tinfo[0] = t;        // threshold bucket
  tinfo[1] = cum;      // total strictly above threshold (< K)
}

__global__ void count_k(const float* __restrict__ keys, int n,
                        const unsigned* tinfo, int* blkA, int* blkE) {
  __shared__ int cA, cE;
  int tid = threadIdx.x;
  if (tid == 0) { cA = 0; cE = 0; }
  __syncthreads();
  int i = blockIdx.x * 256 + tid;
  if (i < n) {
    unsigned b = keybits(keys[i]) >> 20;
    unsigned t = tinfo[0];
    if (b > t) atomicAdd(&cA, 1);
    else if (b == t) atomicAdd(&cE, 1);
  }
  __syncthreads();
  if (tid == 0) { blkA[blockIdx.x] = cA; blkE[blockIdx.x] = cE; }
}

__global__ void scan_k(const int* blkA, const int* blkE, int nb,
                       int* offA, int* offE) {
  if (threadIdx.x | blockIdx.x) return;
  int a = 0, e = 0;
  for (int b = 0; b < nb; ++b) {
    offA[b] = a; a += blkA[b];
    offE[b] = e; e += blkE[b];
  }
  offA[nb] = a; offE[nb] = e;
}

__global__ void write_k(const float* __restrict__ keys, int n,
                        const unsigned* tinfo, const int* offA,
                        const int* offE, int nb, int* perm, int Kw) {
  __shared__ int wA[8], wE[8], baseA[8], baseE[8];
  int tid = threadIdx.x;
  int lane = tid & 31, wv = tid >> 5;
  int i = blockIdx.x * 256 + tid;
  unsigned t = tinfo[0];
  bool va = false, ve = false;
  if (i < n) {
    unsigned b = keybits(keys[i]) >> 20;
    va = (b > t);
    ve = (b == t);
  }
  unsigned long long mA = __ballot(va);
  unsigned long long mE = __ballot(ve);
  unsigned long long below = (1ull << lane) - 1ull;
  int rA = __popcll(mA & below);
  int rE = __popcll(mE & below);
  if (lane == 0) { wA[wv] = (int)__popcll(mA); wE[wv] = (int)__popcll(mE); }
  __syncthreads();
  if (tid == 0) {
    int a = 0, e = 0;
    for (int w = 0; w < 8; ++w) { baseA[w] = a; a += wA[w]; baseE[w] = e; e += wE[w]; }
  }
  __syncthreads();
  int totalAbove = offA[nb];
  if (va) { int pos = offA[blockIdx.x] + baseA[wv] + rA; perm[pos] = i; }
  if (ve) {
    int pos = totalAbove + offE[blockIdx.x] + baseE[wv] + rE;
    if (pos < Kw) perm[pos] = i;
  }
}

// ---------------------------------------------------------------------------
// readout / loss / gating / remap
// ---------------------------------------------------------------------------
__global__ void readout_k(const float* __restrict__ xt,
                          const float* __restrict__ s,
                          const int* __restrict__ perm, int Kw,
                          float* __restrict__ ro) {    // ro[128]
  int col = threadIdx.x;                              // block of 128 threads
  float acc = 0.0f;
  for (int i = blockIdx.x; i < Kw; i += gridDim.x) {
    int p = perm[i];
    acc += xt[(size_t)p * DD + col] * tanhf(s[p]);
  }
  atomicAdd(&ro[col], acc);
}

__global__ void loss_k(const float* ro, float* outLoss) {
  __shared__ float sd[128], sa[128], sb[128];
  int t = threadIdx.x;
  float a = ro[t], b = ro[128 + t];
  sd[t] = a * b; sa[t] = a * a; sb[t] = b * b;
  __syncthreads();
  for (int s = 64; s; s >>= 1) {
    if (t < s) { sd[t] += sd[t + s]; sa[t] += sa[t + s]; sb[t] += sb[t + s]; }
    __syncthreads();
  }
  if (t == 0) outLoss[0] = sd[0] / (sqrtf(sa[0]) * sqrtf(sb[0]) + 1e-8f);
}

__global__ void gate_rows_k(const float* __restrict__ xt,
                            const float* __restrict__ s,
                            const int* __restrict__ perm,
                            float* __restrict__ out, int Kw) {
  int gid = blockIdx.x * blockDim.x + threadIdx.x;
  int i = gid >> 5, lane = gid & 31;
  if (i >= Kw) return;
  int p = perm[i];
  float g = tanhf(s[p]);
  float4 v = ((const float4*)(xt + (size_t)p * DD))[lane];
  v.x *= g; v.y *= g; v.z *= g; v.w *= g;
  ((float4*)(out + (size_t)i * DD))[lane] = v;
}

__global__ void setmap_k(const int* perm, int* nmap, int Kw) {
  int i = blockIdx.x * blockDim.x + threadIdx.x;
  if (i < Kw) nmap[perm[i]] = i;
}

__global__ void edge_filter_k(const int* __restrict__ src,
                              const int* __restrict__ dst,
                              const int* __restrict__ nmap,
                              float* __restrict__ oute, int nE) {
  int e = blockIdx.x * blockDim.x + threadIdx.x;
  if (e >= nE) return;
  int a = nmap[src[e]];
  int b = nmap[dst[e]];
  bool keep = (a >= 0) && (b >= 0);
  oute[e]      = keep ? (float)a : -1.0f;
  oute[nE + e] = keep ? (float)b : -1.0f;
}

__global__ void i2f_k(const int* in, float* out, int n) {
  int i = blockIdx.x * blockDim.x + threadIdx.x;
  if (i < n) out[i] = (float)in[i];
}

__global__ void batch_out_k(const int* batch, const int* perm, float* out, int Kw) {
  int i = blockIdx.x * blockDim.x + threadIdx.x;
  if (i < Kw) out[i] = (float)batch[perm[i]];
}

// ---------------------------------------------------------------------------
// host-side orchestration
// ---------------------------------------------------------------------------
static void run_topk(const float* keys, unsigned* hist, unsigned* tinfo,
                     int* blkA, int* blkE, int* offA, int* offE,
                     int* perm, hipStream_t stream) {
  fill_u32<<<16, 256, 0, stream>>>(hist, 0u, 4096);
  hist_k<<<NB_N, 256, 0, stream>>>(keys, NN, hist);
  thresh_k<<<1, 1, 0, stream>>>(hist, tinfo, KK);
  count_k<<<NB_N, 256, 0, stream>>>(keys, NN, tinfo, blkA, blkE);
  scan_k<<<1, 1, 0, stream>>>(blkA, blkE, NB_N, offA, offE);
  write_k<<<NB_N, 256, 0, stream>>>(keys, NN, tinfo, offA, offE, NB_N, perm, KK);
}

extern "C" void kernel_launch(void* const* d_in, const int* in_sizes, int n_in,
                              void* d_out, int out_size, void* d_ws, size_t ws_size,
                              hipStream_t stream) {
  (void)in_sizes; (void)n_in; (void)out_size; (void)ws_size;

  const float* x       = (const float*)d_in[0];
  const int*   src     = (const int*)d_in[1];
  const int*   dst     = src + EE;
  const int*   batch   = (const int*)d_in[2];
  const float* Wt_rel  = (const float*)d_in[3];
  const float* bt      = (const float*)d_in[4];
  const float* Wt_root = (const float*)d_in[5];
  const float* Wpp_rel = (const float*)d_in[6];
  const float* bpp     = (const float*)d_in[7];
  const float* Wpp_root= (const float*)d_in[8];
  const float* Wnp_rel = (const float*)d_in[9];
  const float* bnp     = (const float*)d_in[10];
  const float* Wnp_root= (const float*)d_in[11];
  const float* Wl_rel  = (const float*)d_in[12];
  const float* bl      = (const float*)d_in[13];
  const float* Wl_root = (const float*)d_in[14];
  const float* Ws_rel  = (const float*)d_in[15];
  const float* bs      = (const float*)d_in[16];
  const float* Ws_root = (const float*)d_in[17];

  // ---- workspace carve-up -------------------------------------------------
  char* w = (char*)d_ws;
  size_t off = 0;
  auto alloc = [&](size_t bytes) -> void* {
    void* p = w + off;
    off = (off + bytes + 255) & ~(size_t)255;
    return p;
  };
  _Float16* wT0   = (_Float16*)alloc(128 * 256 * sizeof(_Float16));
  _Float16* wT1   = (_Float16*)alloc(128 * 256 * sizeof(_Float16));
  _Float16* wT2   = (_Float16*)alloc(128 * 256 * sizeof(_Float16));
  float* agg      = (float*)alloc((size_t)NN * DD * sizeof(float));
  float* xt       = (float*)alloc((size_t)NN * DD * sizeof(float));
  float* xtmp     = (float*)alloc((size_t)NN * DD * sizeof(float));
  float* sagg     = (float*)alloc(NN * sizeof(float));
  float* yrel     = (float*)alloc(NN * sizeof(float));
  float* yroot    = (float*)alloc(NN * sizeof(float));
  float* slp      = (float*)alloc(NN * sizeof(float));
  float* ssp      = (float*)alloc(NN * sizeof(float));
  float* score    = (float*)alloc(NN * sizeof(float));
  unsigned* hist  = (unsigned*)alloc(4096 * sizeof(unsigned));
  unsigned* tinfo = (unsigned*)alloc(2 * sizeof(unsigned));
  int* blkA       = (int*)alloc((NB_N + 1) * sizeof(int));
  int* blkE       = (int*)alloc((NB_N + 1) * sizeof(int));
  int* offA       = (int*)alloc((NB_N + 1) * sizeof(int));
  int* offE       = (int*)alloc((NB_N + 1) * sizeof(int));
  int* permP      = (int*)alloc(KK * sizeof(int));
  int* permM      = (int*)alloc(KK * sizeof(int));
  int* permA      = (int*)alloc(KK * sizeof(int));
  int* nmap       = (int*)alloc(NN * sizeof(int));
  float* ro       = (float*)alloc(256 * sizeof(float));

  float* out_f = (float*)d_out;
  const size_t oX  = 0;
  const size_t oE  = oX + (size_t)KK * DD;
  const size_t oB  = oE + 2 * (size_t)EE;
  const size_t oL  = oB + KK;
  const size_t oPP = oL + 1;
  const size_t oPN = oPP + KK;

  const int gemmBlocks   = (NN + 127) / 128;          // 391
  const int rowsGrid     = (NN * DD + 255) / 256;     // zero agg
  const int edgeWaveGrid = (EE * 32 + 255) / 256;
  const int edgeGrid     = (EE + 255) / 256;
  const int nodeWaveGrid = (NN * 32 + 255) / 256;
  const int kWaveGrid    = (KK * 32 + 255) / 256;
  const int kGrid        = (KK + 255) / 256;

  // ---- weight conversion --------------------------------------------------
  convw_k<<<128, 256, 0, stream>>>(Wt_rel,  Wt_root,  wT0);
  convw_k<<<128, 256, 0, stream>>>(Wpp_rel, Wpp_root, wT1);
  convw_k<<<128, 256, 0, stream>>>(Wnp_rel, Wnp_root, wT2);

  // ---- layer T: x -> x_t --------------------------------------------------
  fill_f32<<<rowsGrid, 256, 0, stream>>>(agg, 0.0f, NN * DD);
  scatter_rows_k<<<edgeWaveGrid, 256, 0, stream>>>(x, src, dst, agg, EE);
  gemm_fused_k<<<gemmBlocks, 256, 0, stream>>>(agg, x, wT0, bt, xt, NN, 1);

  // ---- layer PP: x_t -> x_tl (in xtmp) -----------------------------------
  fill_f32<<<rowsGrid, 256, 0, stream>>>(agg, 0.0f, NN * DD);
  scatter_rows_k<<<edgeWaveGrid, 256, 0, stream>>>(xt, src, dst, agg, EE);
  gemm_fused_k<<<gemmBlocks, 256, 0, stream>>>(agg, xt, wT1, bpp, xtmp, NN, 1);

  // ---- s_lp from x_tl (linearity: segsum(x)@w == segsum(x@w)) -------------
  dots2_k<<<nodeWaveGrid, 256, 0, stream>>>(xtmp, Wl_rel, Wl_root, yrel, yroot, NN);
  fill_f32<<<NB_N, 256, 0, stream>>>(sagg, 0.0f, NN);
  scatter_scalar_k<<<edgeGrid, 256, 0, stream>>>(yrel, src, dst, sagg, EE);
  combine_s_k<<<NB_N, 256, 0, stream>>>(sagg, bl, yroot, slp, NN);

  // ---- layer NP: x_t -> x_ts (reuse xtmp) ---------------------------------
  fill_f32<<<rowsGrid, 256, 0, stream>>>(agg, 0.0f, NN * DD);
  scatter_rows_k<<<edgeWaveGrid, 256, 0, stream>>>(xt, src, dst, agg, EE);
  gemm_fused_k<<<gemmBlocks, 256, 0, stream>>>(agg, xt, wT2, bnp, xtmp, NN, 1);

  // ---- s_sp from x_ts ------------------------------------------------------
  dots2_k<<<nodeWaveGrid, 256, 0, stream>>>(xtmp, Ws_rel, Ws_root, yrel, yroot, NN);
  fill_f32<<<NB_N, 256, 0, stream>>>(sagg, 0.0f, NN);
  scatter_scalar_k<<<edgeGrid, 256, 0, stream>>>(yrel, src, dst, sagg, EE);
  combine_s_k<<<NB_N, 256, 0, stream>>>(sagg, bs, yroot, ssp, NN);

  // ---- score + three top-K selections -------------------------------------
  mean2_k<<<NB_N, 256, 0, stream>>>(slp, ssp, score, NN);
  run_topk(slp,   hist, tinfo, blkA, blkE, offA, offE, permP, stream);
  run_topk(ssp,   hist, tinfo, blkA, blkE, offA, offE, permM, stream);
  run_topk(score, hist, tinfo, blkA, blkE, offA, offE, permA, stream);

  // ---- readouts + cosine loss ---------------------------------------------
  fill_f32<<<1, 256, 0, stream>>>(ro, 0.0f, 256);
  readout_k<<<128, 128, 0, stream>>>(xt, slp, permP, KK, ro);
  readout_k<<<128, 128, 0, stream>>>(xt, ssp, permM, KK, ro + 128);
  loss_k<<<1, 128, 0, stream>>>(ro, out_f + oL);

  // ---- gated x_out ---------------------------------------------------------
  gate_rows_k<<<kWaveGrid, 256, 0, stream>>>(xt, score, permA, out_f + oX, KK);

  // ---- node remap + edge filter -------------------------------------------
  fill_i32<<<NB_N, 256, 0, stream>>>(nmap, -1, NN);
  setmap_k<<<kGrid, 256, 0, stream>>>(permA, nmap, KK);
  edge_filter_k<<<edgeGrid, 256, 0, stream>>>(src, dst, nmap, out_f + oE, EE);

  // ---- remaining outputs ---------------------------------------------------
  batch_out_k<<<kGrid, 256, 0, stream>>>(batch, permA, out_f + oB, KK);
  i2f_k<<<kGrid, 256, 0, stream>>>(permP, out_f + oPP, KK);
  i2f_k<<<kGrid, 256, 0, stream>>>(permM, out_f + oPN, KK);
}
